// LinearAttention_17858474017312
// MI455X (gfx1250) — compile-verified, matrix operand order checked by reading, not run on hardware
//
// CDNA5 / gfx1250 linear-attention block.
// Strategy: f16 WMMA (v_wmma_f32_16x16x32_f16) for all GEMMs, f32 epilogues,
// TDM (tensor_load_to_lds) double-buffered staging for the dominant qkv GEMM.
// This revision: dword-packed A-fragment LDS reads + 2x4 / 2x2 wave tiling
// to cut ds/valu overhead per WMMA.  Workspace requirement: ~392 MB.
#include <hip/hip_runtime.h>
#include <stdint.h>

typedef __attribute__((ext_vector_type(16))) _Float16     v16h;
typedef __attribute__((ext_vector_type(8)))  float        v8f;
typedef __attribute__((ext_vector_type(8)))  unsigned int v8u;
typedef __attribute__((ext_vector_type(4)))  unsigned int u32x4;
typedef __attribute__((ext_vector_type(8)))  int          i32x8;
typedef __attribute__((ext_vector_type(4)))  int          i32x4;

namespace cfg {
constexpr int B = 16, N = 4096, D = 512, H = 8, DH = 64, GD = 16;
constexpr int INNER = 512, QKV = 3 * INNER;   // 1536
constexpr int ROWS = B * N;                   // 65536
constexpr float LN_EPS = 1e-5f;
}

// ---------------- TDM availability probe ----------------------------------
#if defined(__HIP_DEVICE_COMPILE__) && defined(__has_builtin)
#if __has_builtin(__builtin_amdgcn_tensor_load_to_lds) && \
    __has_builtin(__builtin_amdgcn_s_wait_tensorcnt)
#define HAVE_TDM 1
#endif
#endif
#ifndef HAVE_TDM
#define HAVE_TDM 0
#endif

#if HAVE_TDM
// Build a 2D tensor-DMA descriptor (D#) per CDNA5 ISA and
// issue TENSOR_LOAD_TO_LDS.  dim0 = contiguous elements within a row,
// dim1 = rows; stride0 in elements.  data_size_log2: 0=1B,1=2B,2=4B.
__device__ __forceinline__ void tdm_load_2d(uint32_t lds_off, const void* gptr,
                                            uint32_t data_size_log2,
                                            uint32_t tile_d0, uint32_t tile_d1,
                                            uint64_t tensor_d0, uint64_t tensor_d1,
                                            uint64_t stride0) {
  uint64_t ga   = (uint64_t)(uintptr_t)gptr;
  uint64_t g0lo = 1ull /*count=1*/ | ((uint64_t)lds_off << 32);
  uint64_t g0hi = (ga & ((1ull << 57) - 1)) | (2ull << 62);       // type=2
  // group1 bit layout: data_size[17:16], tensor_dim0[79:48],
  // tensor_dim1[111:80], tile_dim0[127:112], tile_dim1[143:128],
  // tensor_dim0_stride[207:160]
  uint64_t w0 = ((uint64_t)data_size_log2 << 16) | ((tensor_d0 & 0xffffull) << 48);
  uint64_t w1 = ((tensor_d0 >> 16) & 0xffffull) |
                ((tensor_d1 & 0xffffffffull) << 16) |
                ((uint64_t)(tile_d0 & 0xffff) << 48);
  uint64_t w2 = (uint64_t)(tile_d1 & 0xffff) | ((stride0 & 0xffffffffull) << 32);
  uint64_t w3 = (stride0 >> 32) & 0xffffull;
  u32x4 g0 = {(uint32_t)g0lo, (uint32_t)(g0lo >> 32),
              (uint32_t)g0hi, (uint32_t)(g0hi >> 32)};
  i32x8 g1 = {(int)(uint32_t)w0, (int)(uint32_t)(w0 >> 32),
              (int)(uint32_t)w1, (int)(uint32_t)(w1 >> 32),
              (int)(uint32_t)w2, (int)(uint32_t)(w2 >> 32),
              (int)(uint32_t)w3, (int)(uint32_t)(w3 >> 32)};
  i32x4 z4 = {0, 0, 0, 0};
#if __clang_major__ >= 23
  i32x8 z8 = {0, 0, 0, 0, 0, 0, 0, 0};
  __builtin_amdgcn_tensor_load_to_lds(g0, g1, z4, z4, z8, 0);
#else
  __builtin_amdgcn_tensor_load_to_lds(g0, g1, z4, z4, 0);
#endif
}
#endif  // HAVE_TDM

// ---------------- WMMA fragment helpers (CDNA5 ISA §7.12.2) ----------
__device__ __forceinline__ v8f wmma_f16(v16h a, v16h b, v8f c) {
  return __builtin_amdgcn_wmma_f32_16x16x32_f16(false, a, false, b, (short)0, c,
                                                false, false);
}
__device__ __forceinline__ int kpair(int e, int lane) {
  return (e < 4 ? 2 * e : 16 + 2 * (e - 4)) + ((lane & 16) ? 8 : 0);
}
// A (16x32 f16) from an fp32 LDS tile, row-major [M][stride]
__device__ __forceinline__ v16h frag_a_f32(const float* t, int m0, int kofs,
                                           int stride, int lane) {
  v16h a;
  int M = m0 + (lane & 15);
#pragma unroll
  for (int e = 0; e < 8; ++e) {
    int kp   = kpair(e, lane);
    a[2 * e]     = (_Float16)t[(size_t)M * stride + kofs + kp];
    a[2 * e + 1] = (_Float16)t[(size_t)M * stride + kofs + kp + 1];
  }
  return a;
}
// A (16x32 f16) from an f16 LDS tile, row-major [M][stride].
// Each K-pair is a contiguous dword-aligned 32-bit word -> ds_load_b32.
__device__ __forceinline__ v16h frag_a_f16(const _Float16* t, int m0, int kofs,
                                           int stride, int lane) {
  union { v16h h; v8u u; } cv;
  int M = m0 + (lane & 15);
#pragma unroll
  for (int e = 0; e < 8; ++e) {
    int kp  = kpair(e, lane);
    cv.u[e] = *(const uint32_t*)(t + (size_t)M * stride + kofs + kp);
  }
  return cv.h;
}
// A (16x32 f16) transposed: logical A[m][k] = t[k][m], tile [K][stride]
__device__ __forceinline__ v16h frag_aT_f16(const _Float16* t, int m0, int kofs,
                                            int stride, int lane) {
  v16h a;
  int M = m0 + (lane & 15);
#pragma unroll
  for (int e = 0; e < 8; ++e) {
    int kp   = kpair(e, lane);
    a[2 * e]     = t[(size_t)(kofs + kp) * stride + M];
    a[2 * e + 1] = t[(size_t)(kofs + kp + 1) * stride + M];
  }
  return a;
}
// B (32x16 f16) from an f16 LDS tile, row-major [K][stride]
__device__ __forceinline__ v16h frag_b_f16(const _Float16* t, int n0, int kofs,
                                           int stride, int lane) {
  v16h b;
  int Nc = n0 + (lane & 15);
#pragma unroll
  for (int e = 0; e < 8; ++e) {
    int kp   = kpair(e, lane);
    b[2 * e]     = t[(size_t)(kofs + kp) * stride + Nc];
    b[2 * e + 1] = t[(size_t)(kofs + kp + 1) * stride + Nc];
  }
  return b;
}

// ---------------- Kernel 1: weight fp32 -> f16 ----------------------------
__global__ __launch_bounds__(256) void cvt_kernel(
    const float* __restrict__ wqkv, const float* __restrict__ wout,
    const float* __restrict__ wg2, _Float16* __restrict__ wqkv16,
    _Float16* __restrict__ wout16, _Float16* __restrict__ wg216) {
  const int stride = gridDim.x * blockDim.x;
  int i = blockIdx.x * blockDim.x + threadIdx.x;
  for (int t = i; t < cfg::D * cfg::QKV; t += stride) wqkv16[t] = (_Float16)wqkv[t];
  for (int t = i; t < cfg::INNER * cfg::D; t += stride) wout16[t] = (_Float16)wout[t];
  for (int t = i; t < cfg::GD * cfg::D; t += stride) wg216[t] = (_Float16)wg2[t];
}

// ---------------- Kernel 2: g1 = x @ w_g1 (rows x 16) ---------------------
__global__ __launch_bounds__(256) void g1_kernel(const float* __restrict__ x,
                                                 const float* __restrict__ wg1,
                                                 float* __restrict__ g1) {
  const int t   = threadIdx.x;
  const int row = blockIdx.x * 16 + (t >> 4);
  const int j   = t & 15;
  const float* xr = x + (size_t)row * cfg::D;
  float acc = 0.f;
  for (int k = 0; k < cfg::D; ++k) acc += xr[k] * wg1[k * cfg::GD + j];
  g1[(size_t)row * cfg::GD + j] = acc;
}

// ---------------- Kernel 3: qkv = silu(x@w_qkv), + LRPE -------------------
// 128x128 output tile per workgroup, K=512 in 16 steps of 32.
// Waves tiled 4x2 over the 8x8 grid of 16x16 C tiles: 2 M-blocks x 4 N-blocks
// each (2 A-frags + 4 B-frags per K-step instead of 1+8).
// TDM double-buffers the x (fp32) and w_qkv (f16) tiles into LDS.
__global__ __launch_bounds__(256) void qkv_kernel(
    const float* __restrict__ x, const _Float16* __restrict__ wq16,
    const float* __restrict__ theta, _Float16* __restrict__ qf,
    _Float16* __restrict__ kf, _Float16* __restrict__ vf) {
  __shared__ float    xa[2][128 * 32];   // 32 KB
  __shared__ _Float16 wb[2][32 * 128];   // 16 KB
  const int r0 = blockIdx.x * 128;
  const int c0 = blockIdx.y * 128;
  const int tid = threadIdx.x, lane = tid & 31, wave = tid >> 5;
  const int wm = wave & 3;   // M-block pair {2wm, 2wm+1}
  const int wn = wave >> 2;  // N-block quad {4wn .. 4wn+3}

  v8f acc[2][4] = {};

#if HAVE_TDM
  if (tid < 32) {  // wave 0 drives the tensor DMA (EXEC ignored by TDM)
    tdm_load_2d((uint32_t)(uintptr_t)&xa[0][0], x + (size_t)r0 * cfg::D, 2, 32,
                128, cfg::D, cfg::ROWS, cfg::D);
    tdm_load_2d((uint32_t)(uintptr_t)&wb[0][0], wq16 + c0, 1, 128, 32, cfg::QKV,
                cfg::D, cfg::QKV);
  }
#endif
  for (int s = 0; s < 16; ++s) {
    const int cur = s & 1;
#if HAVE_TDM
    if (tid < 32) {
      if (s + 1 < 16) {
        const int k1 = (s + 1) * 32, nxt = (s + 1) & 1;
        tdm_load_2d((uint32_t)(uintptr_t)&xa[nxt][0],
                    x + (size_t)r0 * cfg::D + k1, 2, 32, 128, cfg::D, cfg::ROWS,
                    cfg::D);
        tdm_load_2d((uint32_t)(uintptr_t)&wb[nxt][0],
                    wq16 + (size_t)k1 * cfg::QKV + c0, 1, 128, 32, cfg::QKV,
                    cfg::D, cfg::QKV);
        __builtin_amdgcn_s_wait_tensorcnt(2);  // current pair complete
      } else {
        __builtin_amdgcn_s_wait_tensorcnt(0);
      }
    }
    __syncthreads();
#else
    __syncthreads();
    {  // cooperative fallback staging
      const int k0 = s * 32;
      for (int i = tid; i < 1024; i += 256) {  // 128x32 fp32, b128 loads
        int row = i >> 3, c4 = (i & 7) * 4;
        *(float4*)&xa[cur][row * 32 + c4] =
            *(const float4*)(x + (size_t)(r0 + row) * cfg::D + k0 + c4);
      }
      for (int i = tid; i < 512; i += 256) {  // 32x128 f16, b128 loads
        int kk = i >> 4, off = (i & 15) * 8;
        *(uint4*)&wb[cur][kk * 128 + off] =
            *(const uint4*)(wq16 + (size_t)(k0 + kk) * cfg::QKV + c0 + off);
      }
    }
    __syncthreads();
#endif
    v16h a0 = frag_a_f32(&xa[cur][0], (2 * wm + 0) * 16, 0, 32, lane);
    v16h a1 = frag_a_f32(&xa[cur][0], (2 * wm + 1) * 16, 0, 32, lane);
#pragma unroll
    for (int jn = 0; jn < 4; ++jn) {
      v16h b = frag_b_f16(&wb[cur][0], (4 * wn + jn) * 16, 0, 128, lane);
      acc[0][jn] = wmma_f16(a0, b, acc[0][jn]);
      acc[1][jn] = wmma_f16(a1, b, acc[1][jn]);
    }
    __syncthreads();
  }

  // Epilogue: silu, then LRPE(+1/sqrt(2*DH) scale) for q/k, plain store for v
  const float invs = 0.08838834764831845f;  // 1/sqrt(128)
#pragma unroll
  for (int im = 0; im < 2; ++im) {
#pragma unroll
    for (int jn = 0; jn < 4; ++jn) {
#pragma unroll
      for (int r = 0; r < 8; ++r) {
        int row = r0 + (2 * wm + im) * 16 + ((lane & 16) ? 8 + r : r);
        int col = c0 + (4 * wn + jn) * 16 + (lane & 15);
        float val = acc[im][jn][r];
        val = val / (1.f + __expf(-val));  // silu
        int sect = col >> 9, c5 = col & 511;
        int h = c5 >> 6, dh = c5 & 63;
        int bb = row >> 12, n = row & (cfg::N - 1);
        size_t base = (size_t)(bb * cfg::H + h) * cfg::N + n;
        if (sect == 2) {
          vf[base * 64 + dh] = (_Float16)val;
        } else {
          float ang = theta[c5] * (float)n;
          float sn, cs;
          __sincosf(ang, &sn, &cs);
          _Float16* dst = (sect == 0) ? qf : kf;
          dst[base * 128 + dh]      = (_Float16)(val * cs * invs);
          dst[base * 128 + 64 + dh] = (_Float16)(val * sn * invs);
        }
      }
    }
  }
}

// ---------------- Kernel 4: kv[b,h] = k^T v  (128x64, K=N=4096) -----------
// Waves tiled 4x2 over the 8x4 grid of C tiles: 2 M-blocks x 2 N-blocks each.
__global__ __launch_bounds__(256) void kv_kernel(const _Float16* __restrict__ kf,
                                                 const _Float16* __restrict__ vf,
                                                 _Float16* __restrict__ kv16) {
  __shared__ _Float16 kt[32 * 128];  // 8 KB
  __shared__ _Float16 vt[32 * 64];   // 4 KB
  const int bh = blockIdx.x;
  const size_t kbase = (size_t)bh * cfg::N * 128;
  const size_t vbase = (size_t)bh * cfg::N * 64;
  const int tid = threadIdx.x, lane = tid & 31, wave = tid >> 5;
  const int wm = wave & 3, wn = wave >> 2;
  v8f acc[2][2] = {};
  for (int s = 0; s < cfg::N / 32; ++s) {
    const int n0 = s * 32;
    __syncthreads();
    {  // contiguous tiles -> b128 loads
      const uint4* ks = (const uint4*)(kf + kbase + (size_t)n0 * 128);
      ((uint4*)kt)[tid]       = ks[tid];
      ((uint4*)kt)[tid + 256] = ks[tid + 256];
      ((uint4*)vt)[tid] = ((const uint4*)(vf + vbase + (size_t)n0 * 64))[tid];
    }
    __syncthreads();
    v16h a0 = frag_aT_f16(kt, (2 * wm + 0) * 16, 0, 128, lane);  // A = k^T
    v16h a1 = frag_aT_f16(kt, (2 * wm + 1) * 16, 0, 128, lane);
#pragma unroll
    for (int jn = 0; jn < 2; ++jn) {
      v16h b = frag_b_f16(vt, (2 * wn + jn) * 16, 0, 64, lane);
      acc[0][jn] = wmma_f16(a0, b, acc[0][jn]);
      acc[1][jn] = wmma_f16(a1, b, acc[1][jn]);
    }
  }
#pragma unroll
  for (int im = 0; im < 2; ++im) {
#pragma unroll
    for (int jn = 0; jn < 2; ++jn) {
#pragma unroll
      for (int r = 0; r < 8; ++r) {
        int m   = (2 * wm + im) * 16 + ((lane & 16) ? 8 + r : r);
        int col = (2 * wn + jn) * 16 + (lane & 15);
        kv16[(size_t)bh * 8192 + m * 64 + col] = (_Float16)acc[im][jn][r];
      }
    }
  }
}

// ---------------- Kernel 5: out = q @ kv, scatter heads -------------------
__global__ __launch_bounds__(256) void attn_out_kernel(
    const _Float16* __restrict__ qf, const _Float16* __restrict__ kv16,
    _Float16* __restrict__ attn) {
  __shared__ _Float16 qt[128 * 128];  // 32 KB
  __shared__ _Float16 kvt[128 * 64];  // 16 KB
  const int bh = blockIdx.x;
  const int n0 = blockIdx.y * 128;
  const int tid = threadIdx.x, lane = tid & 31, wave = tid >> 5;
  const int wm = wave & 3, wn = wave >> 2;
  {
    const uint4* qs = (const uint4*)(qf + ((size_t)bh * cfg::N + n0) * 128);
#pragma unroll
    for (int i = 0; i < 8; ++i) ((uint4*)qt)[tid + 256 * i] = qs[tid + 256 * i];
    const uint4* kvs = (const uint4*)(kv16 + (size_t)bh * 8192);
#pragma unroll
    for (int i = 0; i < 4; ++i) ((uint4*)kvt)[tid + 256 * i] = kvs[tid + 256 * i];
  }
  __syncthreads();
  v8f acc[2][2] = {};
#pragma unroll
  for (int ks = 0; ks < 4; ++ks) {  // K = 128
    v16h a0 = frag_a_f16(qt, (2 * wm + 0) * 16, ks * 32, 128, lane);
    v16h a1 = frag_a_f16(qt, (2 * wm + 1) * 16, ks * 32, 128, lane);
#pragma unroll
    for (int jn = 0; jn < 2; ++jn) {
      v16h b = frag_b_f16(kvt, (2 * wn + jn) * 16, ks * 32, 64, lane);
      acc[0][jn] = wmma_f16(a0, b, acc[0][jn]);
      acc[1][jn] = wmma_f16(a1, b, acc[1][jn]);
    }
  }
  const int bb = bh >> 3, h = bh & 7;
#pragma unroll
  for (int im = 0; im < 2; ++im) {
#pragma unroll
    for (int jn = 0; jn < 2; ++jn) {
#pragma unroll
      for (int r = 0; r < 8; ++r) {
        int n   = n0 + (2 * wm + im) * 16 + ((lane & 16) ? 8 + r : r);
        int col = (2 * wn + jn) * 16 + (lane & 15);
        attn[((size_t)bb * cfg::N + n) * cfg::INNER + h * 64 + col] =
            (_Float16)acc[im][jn][r];
      }
    }
  }
}

// ---------------- Kernel 6: LN + gate + @w_out + b_out --------------------
__global__ __launch_bounds__(256) void final_kernel(
    const _Float16* __restrict__ attn, const float* __restrict__ g1,
    const _Float16* __restrict__ wg216, const _Float16* __restrict__ wout16,
    const float* __restrict__ gamma, const float* __restrict__ beta,
    const float* __restrict__ bout, float* __restrict__ out) {
  __shared__ float    mu[128], rs[128];
  __shared__ float    g1s[128 * 16];   // 8 KB
  __shared__ _Float16 At[128 * 32];    // 8 KB
  __shared__ _Float16 Bt[32 * 128];    // 8 KB
  __shared__ _Float16 g2s[16 * 32];    // 1 KB
  const int r0 = blockIdx.x * 128;
  const int c0 = blockIdx.y * 128;
  const int tid = threadIdx.x, lane = tid & 31, wave = tid >> 5;
  const int wm = wave & 3, wn = wave >> 2;

  if (tid < 128) {  // per-row LayerNorm statistics (f32)
    const uint4* rp = (const uint4*)(attn + (size_t)(r0 + tid) * cfg::INNER);
    float s = 0.f, s2 = 0.f;
    for (int i = 0; i < 64; ++i) {
      union { uint4 q; _Float16 h[8]; } cv;
      cv.q = rp[i];
#pragma unroll
      for (int t = 0; t < 8; ++t) {
        float v = (float)cv.h[t];
        s += v;
        s2 += v * v;
      }
    }
    float m = s * (1.f / 512.f);
    mu[tid] = m;
    rs[tid] = rsqrtf(s2 * (1.f / 512.f) - m * m + cfg::LN_EPS);
  }
  for (int i = tid; i < 128 * 16; i += 256) g1s[i] = g1[(size_t)r0 * 16 + i];
  __syncthreads();

  v8f acc[2][4] = {};
  for (int s = 0; s < 16; ++s) {
    const int k0 = s * 32;
    __syncthreads();
    if (tid < 128) {  // stage w_g2 chunk [16][32]
      int j = tid >> 3, kk = (tid & 7) * 4;
#pragma unroll
      for (int t = 0; t < 4; ++t)
        g2s[j * 32 + kk + t] = wg216[j * cfg::D + k0 + kk + t];
    }
    for (int i = tid; i < 512; i += 256) {  // stage w_out chunk [32][128]
      int kk = i >> 4, off = (i & 15) * 8;
      *(uint4*)&Bt[kk * 128 + off] =
          *(const uint4*)(wout16 + (size_t)(k0 + kk) * cfg::D + c0 + off);
    }
    if (tid < 32 && s + 1 < 16)  // warm L2/L0 for the next weight chunk
      __builtin_prefetch(wout16 + (size_t)(k0 + 32 + tid) * cfg::D + c0, 0, 1);
    __syncthreads();
    for (int i = tid; i < 128 * 32; i += 256) {  // A = LN(out)*gate, f16
      int r = i >> 5, kk = i & 31, k = k0 + kk;
      float v  = (float)attn[(size_t)(r0 + r) * cfg::INNER + k];
      float ln = (v - mu[r]) * rs[r] * gamma[k] + beta[k];
      float gd = 0.f;
#pragma unroll
      for (int j = 0; j < 16; ++j) gd += g1s[r * 16 + j] * (float)g2s[j * 32 + kk];
      float gate = 1.f / (1.f + __expf(-gd));
      At[i] = (_Float16)(ln * gate);
    }
    __syncthreads();
    v16h a0 = frag_a_f16(At, (2 * wm + 0) * 16, 0, 32, lane);
    v16h a1 = frag_a_f16(At, (2 * wm + 1) * 16, 0, 32, lane);
#pragma unroll
    for (int jn = 0; jn < 4; ++jn) {
      v16h b = frag_b_f16(Bt, (4 * wn + jn) * 16, 0, 128, lane);
      acc[0][jn] = wmma_f16(a0, b, acc[0][jn]);
      acc[1][jn] = wmma_f16(a1, b, acc[1][jn]);
    }
  }
#pragma unroll
  for (int im = 0; im < 2; ++im) {
#pragma unroll
    for (int jn = 0; jn < 4; ++jn) {
#pragma unroll
      for (int r = 0; r < 8; ++r) {
        int row = r0 + (2 * wm + im) * 16 + ((lane & 16) ? 8 + r : r);
        int col = c0 + (4 * wn + jn) * 16 + (lane & 15);
        out[(size_t)row * cfg::D + col] = acc[im][jn][r] + bout[col];
      }
    }
  }
}

// ---------------- Host launcher -------------------------------------------
extern "C" void kernel_launch(void* const* d_in, const int* in_sizes, int n_in,
                              void* d_out, int out_size, void* d_ws,
                              size_t ws_size, hipStream_t stream) {
  (void)in_sizes; (void)n_in; (void)out_size; (void)ws_size;
  const float* x     = (const float*)d_in[0];
  const float* wqkv  = (const float*)d_in[1];
  const float* theta = (const float*)d_in[2];
  const float* gamma = (const float*)d_in[3];
  const float* beta  = (const float*)d_in[4];
  const float* wg1   = (const float*)d_in[5];
  const float* wg2   = (const float*)d_in[6];
  const float* wout  = (const float*)d_in[7];
  const float* bout  = (const float*)d_in[8];
  float* out = (float*)d_out;

  char* p = (char*)d_ws;
  _Float16* wqkv16 = (_Float16*)p;  p += (size_t)cfg::D * cfg::QKV * 2;      // 1.5 MB
  _Float16* wout16 = (_Float16*)p;  p += (size_t)cfg::INNER * cfg::D * 2;    // 0.5 MB
  _Float16* wg216  = (_Float16*)p;  p += (size_t)cfg::GD * cfg::D * 2;       // 16 KB
  _Float16* qf     = (_Float16*)p;  p += (size_t)cfg::B * cfg::H * cfg::N * 128 * 2;  // 128 MB
  _Float16* kf     = (_Float16*)p;  p += (size_t)cfg::B * cfg::H * cfg::N * 128 * 2;  // 128 MB
  _Float16* vf     = (_Float16*)p;  p += (size_t)cfg::B * cfg::H * cfg::N * 64 * 2;   // 64 MB
  _Float16* kv16   = (_Float16*)p;  p += (size_t)cfg::B * cfg::H * 128 * 64 * 2;      // 2 MB
  _Float16* attn   = (_Float16*)p;  p += (size_t)cfg::ROWS * cfg::INNER * 2;          // 64 MB
  float*    g1buf  = (float*)p;     p += (size_t)cfg::ROWS * cfg::GD * 4;             // 4 MB

  cvt_kernel<<<3072, 256, 0, stream>>>(wqkv, wout, wg2, wqkv16, wout16, wg216);
  g1_kernel<<<cfg::ROWS / 16, 256, 0, stream>>>(x, wg1, g1buf);
  qkv_kernel<<<dim3(cfg::ROWS / 128, cfg::QKV / 128), 256, 0, stream>>>(
      x, wqkv16, theta, qf, kf, vf);
  kv_kernel<<<cfg::B * cfg::H, 256, 0, stream>>>(kf, vf, kv16);
  attn_out_kernel<<<dim3(cfg::B * cfg::H, cfg::N / 128), 256, 0, stream>>>(
      qf, kv16, attn);
  final_kernel<<<dim3(cfg::ROWS / 128, cfg::D / 128), 256, 0, stream>>>(
      attn, g1buf, wg216, wout16, gamma, beta, bout, out);
}
